// MultiheadAttention_59150289601069
// MI455X (gfx1250) — matrix operand order repaired and checked by reading, not compiled, not run on hardware
//
#include <hip/hip_runtime.h>
#include <hip/hip_bf16.h>
#include <math.h>

#define NH 8
#define HD 64
#define DMODEL 512
#define PS 16
#define GWW 44
#define S_TOK 704
#define NVIEW 6
#define LMK 16
#define TKV 3520
#define IH 256
#define IW 704
#define LDSK 40         // padded LDS stride (32 k + 8 pad) in 16-bit elements
#define PE_K 65536
#define PE_CHUNKS 16
#define PE_KCHUNK (PE_K / PE_CHUNKS)   // 4096 -> 128 WMMA K-steps per block

typedef __attribute__((ext_vector_type(16))) __bf16 bf16x16;
typedef __attribute__((ext_vector_type(8)))  __bf16 bf16x8v;
typedef __attribute__((ext_vector_type(8)))  float  f32x8;
typedef __attribute__((ext_vector_type(4)))  float  f32x4;
typedef __attribute__((ext_vector_type(8)))  unsigned short u16x8;

// Native packed f32->bf16 (RNE): lowers to v_cvt_pk_bf16_f32 on gfx1250.
__device__ __forceinline__ u16x8 cvt8(f32x4 a, f32x4 b) {
    f32x8 f = __builtin_shufflevector(a, b, 0, 1, 2, 3, 4, 5, 6, 7);
    bf16x8v h = __builtin_convertvector(f, bf16x8v);
    union { bf16x8v h; u16x8 v; } u;
    u.h = h;
    return u.v;
}

// A fragment: 16x32 bf16 tile in LDS (row-major, stride LDSK).
// lane 0-15: M=lane, K in {0..7,16..23}; lane 16-31: M=lane-16, K in {8..15,24..31}
__device__ __forceinline__ bf16x16 frag_a(const unsigned short* Alds, int mbase, int lane) {
    int row = mbase + (lane & 15);
    int kh  = (lane >> 4) << 3;
    const unsigned short* p = Alds + row * LDSK + kh;
    union { bf16x16 v; u16x8 h[2]; } u;
    u.h[0] = *(const u16x8*)(p);
    u.h[1] = *(const u16x8*)(p + 16);
    return u.v;
}

// B fragment: 32x16 bf16; LDS tile stored (N,K) row-major (stride LDSK).
// lane 0-15: N=lane, K in {0..7,16..23}; lane 16-31: N=lane-16, K in {8..15,24..31}
__device__ __forceinline__ bf16x16 frag_b(const unsigned short* Blds, int nbase, int lane) {
    int col = nbase + (lane & 15);
    int kh  = (lane >> 4) << 3;
    const unsigned short* p = Blds + col * LDSK + kh;
    union { bf16x16 v; u16x8 h[2]; } u;
    u.h[0] = *(const u16x8*)(p);
    u.h[1] = *(const u16x8*)(p + 16);
    return u.v;
}

// ---------------------------------------------------------------------------
// Patch-embed conv as WMMA GEMM with double-buffered LDS.
// sel_mode==0: per-view (blockIdx.z = view), full K, store (acc+bias)*mask.
// sel_mode==1: PE conv split-K (blockIdx.z = K-chunk), view=*feat_tag,
//              store raw partial tile to out + chunk*S_TOK*DMODEL.
// ---------------------------------------------------------------------------
__global__ void __launch_bounds__(256)
conv_gemm_kernel(const float* __restrict__ img, const float* __restrict__ Wt,
                 const float* __restrict__ bias, const float* __restrict__ maskf,
                 float* __restrict__ out, int K, int kchunk, int sel_mode,
                 const int* __restrict__ feat_tag)
{
    __shared__ __align__(16) unsigned short Alds[2][64 * LDSK];
    __shared__ __align__(16) unsigned short Blds[2][128 * LDSK];

    const int tid  = threadIdx.x;
    const int lane = tid & 31;
    const int wave = tid >> 5;
    const int wm = wave >> 2, wn = wave & 3;
    const int m0 = blockIdx.x * 64, n0 = blockIdx.y * 128;

    int view, kb, ke;
    if (sel_mode) { view = *feat_tag; kb = blockIdx.z * kchunk; ke = kb + kchunk; }
    else          { view = blockIdx.z; kb = 0; ke = K; }
    float* dst = out + (size_t)blockIdx.z * S_TOK * DMODEL;
    const float* src = img + (size_t)view * (size_t)K * IW;

    const int arow = tid >> 2, apart = (tid & 3) * 8;
    const int s_tok = m0 + arow;
    const int gh = s_tok / GWW, gw = s_tok % GWW;
    const int brow = tid >> 1, bpart = (tid & 1) * 16;

    f32x4 ra0, ra1, rb0, rb1, rb2, rb3;

    auto gload = [&](int k0) {
        int k = k0 + apart;                    // k = c*256 + py*16 + px
        int c = k >> 8, rem = k & 255, py = rem >> 4, px = rem & 15;
        const float* p = src + (size_t)(c * IH + gh * PS + py) * IW + gw * PS + px;
        ra0 = *(const f32x4*)p; ra1 = *(const f32x4*)(p + 4);
        const float* q = Wt + (size_t)(n0 + brow) * K + k0 + bpart;
        rb0 = *(const f32x4*)q;       rb1 = *(const f32x4*)(q + 4);
        rb2 = *(const f32x4*)(q + 8); rb3 = *(const f32x4*)(q + 12);
        __builtin_prefetch(p + 4 * IW, 0, 1);
        __builtin_prefetch(q + 64, 0, 1);
    };
    auto lstore = [&](int b) {
        *(u16x8*)(Alds[b] + arow * LDSK + apart) = cvt8(ra0, ra1);
        unsigned short* q = Blds[b] + brow * LDSK + bpart;
        *(u16x8*)q       = cvt8(rb0, rb1);
        *(u16x8*)(q + 8) = cvt8(rb2, rb3);
    };

    f32x8 acc[2][2] = {};
    gload(kb);
    lstore(0);
    __syncthreads();

    int buf = 0;
    for (int k0 = kb; k0 < ke; k0 += 32) {
        const bool more = (k0 + 32) < ke;
        if (more) gload(k0 + 32);              // next tile's loads in flight
        bf16x16 fa0 = frag_a(Alds[buf], wm * 32,      lane);
        bf16x16 fa1 = frag_a(Alds[buf], wm * 32 + 16, lane);
        bf16x16 fb0 = frag_b(Blds[buf], wn * 32,      lane);
        bf16x16 fb1 = frag_b(Blds[buf], wn * 32 + 16, lane);
        acc[0][0] = __builtin_amdgcn_wmma_f32_16x16x32_bf16(false, fa0, false, fb0, (short)0, acc[0][0], false, false);
        acc[0][1] = __builtin_amdgcn_wmma_f32_16x16x32_bf16(false, fa0, false, fb1, (short)0, acc[0][1], false, false);
        acc[1][0] = __builtin_amdgcn_wmma_f32_16x16x32_bf16(false, fa1, false, fb0, (short)0, acc[1][0], false, false);
        acc[1][1] = __builtin_amdgcn_wmma_f32_16x16x32_bf16(false, fa1, false, fb1, (short)0, acc[1][1], false, false);
        if (more) lstore(buf ^ 1);             // fill other buffer
        __syncthreads();
        buf ^= 1;
    }

    const int cn = lane & 15;
    const int mo = (lane >> 4) * 8;
    #pragma unroll
    for (int i = 0; i < 2; ++i)
      #pragma unroll
      for (int j = 0; j < 2; ++j) {
        int col = n0 + wn * 32 + j * 16 + cn;
        float bv = sel_mode ? 0.f : bias[col];
        #pragma unroll
        for (int r = 0; r < 8; ++r) {
            int row = m0 + wm * 32 + i * 16 + mo + r;
            float v = acc[i][j][r];
            dst[(size_t)row * DMODEL + col] = sel_mode ? v : (v + bv) * maskf[row];
        }
      }
}

// ---------------------------------------------------------------------------
// Generic GEMM: C[m,n] = sum_k A[m,k]*W[n,k] + bias[n]; A row-major (lda=K).
// Double-buffered LDS, same WMMA micro-tiling.
// ---------------------------------------------------------------------------
__global__ void __launch_bounds__(256)
gemm_nt_kernel(const float* __restrict__ A, const float* __restrict__ Wt,
               const float* __restrict__ bias, float* __restrict__ C,
               int K, int ldc)
{
    __shared__ __align__(16) unsigned short Alds[2][64 * LDSK];
    __shared__ __align__(16) unsigned short Blds[2][128 * LDSK];

    const int tid  = threadIdx.x;
    const int lane = tid & 31;
    const int wave = tid >> 5;
    const int wm = wave >> 2, wn = wave & 3;
    const int m0 = blockIdx.x * 64, n0 = blockIdx.y * 128;

    const int arow = tid >> 2, apart = (tid & 3) * 8;
    const int brow = tid >> 1, bpart = (tid & 1) * 16;

    f32x4 ra0, ra1, rb0, rb1, rb2, rb3;

    auto gload = [&](int k0) {
        const float* p = A + (size_t)(m0 + arow) * K + k0 + apart;
        ra0 = *(const f32x4*)p; ra1 = *(const f32x4*)(p + 4);
        const float* q = Wt + (size_t)(n0 + brow) * K + k0 + bpart;
        rb0 = *(const f32x4*)q;       rb1 = *(const f32x4*)(q + 4);
        rb2 = *(const f32x4*)(q + 8); rb3 = *(const f32x4*)(q + 12);
        __builtin_prefetch(q + 64, 0, 1);
    };
    auto lstore = [&](int b) {
        *(u16x8*)(Alds[b] + arow * LDSK + apart) = cvt8(ra0, ra1);
        unsigned short* q = Blds[b] + brow * LDSK + bpart;
        *(u16x8*)q       = cvt8(rb0, rb1);
        *(u16x8*)(q + 8) = cvt8(rb2, rb3);
    };

    f32x8 acc[2][2] = {};
    gload(0);
    lstore(0);
    __syncthreads();

    int buf = 0;
    for (int k0 = 0; k0 < K; k0 += 32) {
        const bool more = (k0 + 32) < K;
        if (more) gload(k0 + 32);
        bf16x16 fa0 = frag_a(Alds[buf], wm * 32,      lane);
        bf16x16 fa1 = frag_a(Alds[buf], wm * 32 + 16, lane);
        bf16x16 fb0 = frag_b(Blds[buf], wn * 32,      lane);
        bf16x16 fb1 = frag_b(Blds[buf], wn * 32 + 16, lane);
        acc[0][0] = __builtin_amdgcn_wmma_f32_16x16x32_bf16(false, fa0, false, fb0, (short)0, acc[0][0], false, false);
        acc[0][1] = __builtin_amdgcn_wmma_f32_16x16x32_bf16(false, fa0, false, fb1, (short)0, acc[0][1], false, false);
        acc[1][0] = __builtin_amdgcn_wmma_f32_16x16x32_bf16(false, fa1, false, fb0, (short)0, acc[1][0], false, false);
        acc[1][1] = __builtin_amdgcn_wmma_f32_16x16x32_bf16(false, fa1, false, fb1, (short)0, acc[1][1], false, false);
        if (more) lstore(buf ^ 1);
        __syncthreads();
        buf ^= 1;
    }

    const int cn = lane & 15;
    const int mo = (lane >> 4) * 8;
    #pragma unroll
    for (int i = 0; i < 2; ++i)
      #pragma unroll
      for (int j = 0; j < 2; ++j) {
        int col = n0 + wn * 32 + j * 16 + cn;
        float bv = bias[col];
        #pragma unroll
        for (int r = 0; r < 8; ++r) {
            int row = m0 + wm * 32 + i * 16 + mo + r;
            C[(size_t)row * ldc + col] = acc[i][j][r] + bv;
        }
      }
}

// Reduce PE split-K partials + bias + mask -> petok
__global__ void reduce_pe_kernel(const float* __restrict__ partials,
                                 const float* __restrict__ bias,
                                 const float* __restrict__ maskf,
                                 float* __restrict__ petok) {
    int idx = blockIdx.x * blockDim.x + threadIdx.x;
    if (idx >= S_TOK * DMODEL) return;
    int srow = idx / DMODEL, col = idx % DMODEL;
    float s = 0.f;
    #pragma unroll
    for (int z = 0; z < PE_CHUNKS; ++z)
        s += partials[(size_t)z * S_TOK * DMODEL + idx];
    petok[idx] = (s + bias[col]) * maskf[srow];
}

// --------------------------- small helper kernels --------------------------

__global__ void mask_kernel(const float* __restrict__ mask_map, float* __restrict__ maskf) {
    int s = blockIdx.x * blockDim.x + threadIdx.x;
    if (s >= S_TOK) return;
    int gh = s / GWW, gw = s % GWW;
    float sum = 0.f;
    for (int py = 0; py < PS; ++py)
        for (int px = 0; px < PS; ++px)
            sum += mask_map[(gh * PS + py) * IW + gw * PS + px];
    maskf[s] = (sum * (1.f / 256.f) > 0.f) ? 1.f : 0.f;
}

__global__ void prep_qn_kernel(const float* __restrict__ qkv, const float* __restrict__ petok,
                               const int* __restrict__ feat_tag, float* __restrict__ qn) {
    int idx = blockIdx.x * blockDim.x + threadIdx.x;
    if (idx >= NH * S_TOK) return;
    int h = idx / S_TOK, s = idx % S_TOK;
    int ft = *feat_tag;
    const float* q  = qkv + ((size_t)ft * S_TOK + s) * (3 * DMODEL) + h * (3 * HD);
    const float* pe = petok + (size_t)s * DMODEL + h * HD;
    float v[HD]; float ss = 0.f;
    #pragma unroll
    for (int d = 0; d < HD; ++d) { v[d] = q[d] + pe[d]; ss += v[d] * v[d]; }
    float inv = 1.f / fmaxf(sqrtf(ss), 1e-12f);
    float* dstp = qn + ((size_t)h * S_TOK + s) * HD;
    #pragma unroll
    for (int d = 0; d < HD; ++d) dstp[d] = v[d] * inv;
}

__global__ void prep_kn_kernel(const float* __restrict__ qkv, const int* __restrict__ feat_tag,
                               float* __restrict__ kn, float* __restrict__ vc) {
    int idx = blockIdx.x * blockDim.x + threadIdx.x;
    if (idx >= NH * TKV) return;
    int h = idx / TKV, t = idx % TKV;
    int ft = *feat_tag;
    int j = t / S_TOK, s = t % S_TOK;
    int view = j + (j >= ft ? 1 : 0);
    const float* kp = qkv + ((size_t)view * S_TOK + s) * (3 * DMODEL) + h * (3 * HD) + HD;
    const float* vp = kp + HD;
    float v[HD]; float ss = 0.f;
    #pragma unroll
    for (int d = 0; d < HD; ++d) { v[d] = kp[d]; ss += v[d] * v[d]; }
    float inv = 1.f / fmaxf(sqrtf(ss), 1e-12f);
    float* kd = kn + ((size_t)h * TKV + t) * HD;
    float* vd = vc + ((size_t)h * TKV + t) * HD;
    #pragma unroll
    for (int d = 0; d < HD; ++d) { kd[d] = v[d] * inv; vd[d] = vp[d]; }
}

__global__ void landmark_kernel(const float* __restrict__ src, float* __restrict__ dst,
                                int bin, int Tot) {
    int idx = blockIdx.x * blockDim.x + threadIdx.x;
    if (idx >= NH * LMK * HD) return;
    int h = idx / (LMK * HD), l = (idx / HD) % LMK, d = idx % HD;
    const float* p = src + ((size_t)h * Tot + (size_t)l * bin) * HD + d;
    float s = 0.f;
    for (int r = 0; r < bin; ++r) s += p[(size_t)r * HD];
    dst[idx] = s / (float)bin;
}

__global__ void attn1_kernel(const float* __restrict__ qn, const float* __restrict__ kL,
                             float* __restrict__ a1) {
    int idx = blockIdx.x * blockDim.x + threadIdx.x;
    if (idx >= NH * S_TOK) return;
    int h = idx / S_TOK;
    const float* q  = qn + (size_t)idx * HD;
    const float* kl = kL + (size_t)h * LMK * HD;
    float sc[LMK]; float mx = -1e30f;
    for (int l = 0; l < LMK; ++l) {
        float dot = 0.f;
        #pragma unroll
        for (int d = 0; d < HD; ++d) dot += q[d] * kl[(size_t)l * HD + d];
        sc[l] = dot * 0.125f;
        mx = fmaxf(mx, sc[l]);
    }
    float sum = 0.f;
    for (int l = 0; l < LMK; ++l) { sc[l] = __expf(sc[l] - mx); sum += sc[l]; }
    float inv = 1.f / sum;
    float* dstp = a1 + (size_t)idx * LMK;
    for (int l = 0; l < LMK; ++l) dstp[l] = sc[l] * inv;
}

__global__ void __launch_bounds__(256)
attn2_kernel(const float* __restrict__ qL, const float* __restrict__ kn,
             const float* __restrict__ vc, float* __restrict__ sc,
             float* __restrict__ a2v) {
    __shared__ float qrow[HD];
    __shared__ float red[256];
    const int tid = threadIdx.x;
    const int h = blockIdx.x >> 4;
    const int l = blockIdx.x & 15;
    if (tid < HD) qrow[tid] = qL[((size_t)h * LMK + l) * HD + tid];
    __syncthreads();

    float* srow = sc + ((size_t)h * LMK + l) * TKV;
    const float* kbase = kn + (size_t)h * TKV * HD;
    float lmax = -1e30f;
    for (int t = tid; t < TKV; t += 256) {
        const float* kp = kbase + (size_t)t * HD;
        float dot = 0.f;
        #pragma unroll
        for (int d = 0; d < HD; ++d) dot += qrow[d] * kp[d];
        dot *= 0.125f;
        srow[t] = dot;
        lmax = fmaxf(lmax, dot);
    }
    red[tid] = lmax; __syncthreads();
    for (int off = 128; off; off >>= 1) {
        if (tid < off) red[tid] = fmaxf(red[tid], red[tid + off]);
        __syncthreads();
    }
    float gmax = red[0]; __syncthreads();

    float lsum = 0.f;
    for (int t = tid; t < TKV; t += 256) {
        float w = __expf(srow[t] - gmax);
        srow[t] = w;
        lsum += w;
    }
    red[tid] = lsum; __syncthreads();
    for (int off = 128; off; off >>= 1) {
        if (tid < off) red[tid] += red[tid + off];
        __syncthreads();
    }
    float inv = 1.f / red[0];
    __syncthreads();

    const int d = tid & 63, grp = tid >> 6;
    const float* vbase = vc + (size_t)h * TKV * HD + d;
    float acc = 0.f;
    for (int t = grp; t < TKV; t += 4) acc += srow[t] * vbase[(size_t)t * HD];
    red[tid] = acc; __syncthreads();
    if (grp == 0)
        a2v[((size_t)h * LMK + l) * HD + d] =
            (red[d] + red[64 + d] + red[128 + d] + red[192 + d]) * inv;
}

__global__ void attn3_kernel(const float* __restrict__ a1, const float* __restrict__ a2v,
                             float* __restrict__ vals) {
    int s = blockIdx.x;
    int h = threadIdx.x >> 6, d = threadIdx.x & 63;
    const float* arow = a1 + ((size_t)h * S_TOK + s) * LMK;
    const float* bm   = a2v + (size_t)h * LMK * HD + d;
    float acc = 0.f;
    #pragma unroll
    for (int l = 0; l < LMK; ++l) acc += arow[l] * bm[(size_t)l * HD];
    vals[(size_t)s * DMODEL + h * HD + d] = acc;
}

// ---------------------------------------------------------------------------

extern "C" void kernel_launch(void* const* d_in, const int* in_sizes, int n_in,
                              void* d_out, int out_size, void* d_ws, size_t ws_size,
                              hipStream_t stream) {
    const float* x         = (const float*)d_in[0];
    const float* pe        = (const float*)d_in[1];
    const float* mask_map  = (const float*)d_in[2];
    const float* conv_in_w = (const float*)d_in[3];
    const float* conv_in_b = (const float*)d_in[4];
    const float* conv_pe_w = (const float*)d_in[5];
    const float* conv_pe_b = (const float*)d_in[6];
    const float* qkv_w     = (const float*)d_in[7];
    const float* qkv_b     = (const float*)d_in[8];
    const float* o_w       = (const float*)d_in[9];
    const float* o_b       = (const float*)d_in[10];
    const int*   feat_tag  = (const int*)d_in[11];
    float* out = (float*)d_out;

    float* ws = (float*)d_ws;
    size_t off = 0;
    float* maskf   = ws + off; off += 768;
    float* tokens  = ws + off; off += (size_t)NVIEW * S_TOK * DMODEL;
    float* pe_part = ws + off; off += (size_t)PE_CHUNKS * S_TOK * DMODEL;
    float* petok   = ws + off; off += (size_t)S_TOK * DMODEL;
    float* qkvb    = ws + off; off += (size_t)NVIEW * S_TOK * 3 * DMODEL;
    float* qn      = ws + off; off += (size_t)NH * S_TOK * HD;
    float* kn      = ws + off; off += (size_t)NH * TKV * HD;
    float* vc      = ws + off; off += (size_t)NH * TKV * HD;
    float* qL      = ws + off; off += (size_t)NH * LMK * HD;
    float* kL      = ws + off; off += (size_t)NH * LMK * HD;
    float* a1      = ws + off; off += (size_t)NH * S_TOK * LMK;
    float* sc      = ws + off; off += (size_t)NH * LMK * TKV;
    float* a2v     = ws + off; off += (size_t)NH * LMK * HD;
    float* vals    = ws + off; off += (size_t)S_TOK * DMODEL;

    mask_kernel<<<3, 256, 0, stream>>>(mask_map, maskf);

    // x patch embed: 6 views, K = 3*256 = 768 -> 264 blocks
    conv_gemm_kernel<<<dim3(11, 4, 6), 256, 0, stream>>>(
        x, conv_in_w, conv_in_b, maskf, tokens, 768, 768, 0, feat_tag);

    // pe patch embed (view = feat_tag), K = 65536, split-K x16 -> 704 blocks
    conv_gemm_kernel<<<dim3(11, 4, PE_CHUNKS), 256, 0, stream>>>(
        pe, conv_pe_w, conv_pe_b, maskf, pe_part, PE_K, PE_KCHUNK, 1, feat_tag);
    reduce_pe_kernel<<<(S_TOK * DMODEL + 255) / 256, 256, 0, stream>>>(
        pe_part, conv_pe_b, maskf, petok);

    // QKV: (6*704, 512) x (1536, 512)^T -> 792 blocks
    gemm_nt_kernel<<<dim3(66, 12), 256, 0, stream>>>(
        tokens, qkv_w, qkv_b, qkvb, DMODEL, 3 * DMODEL);

    prep_qn_kernel<<<22, 256, 0, stream>>>(qkvb, petok, feat_tag, qn);
    prep_kn_kernel<<<110, 256, 0, stream>>>(qkvb, feat_tag, kn, vc);
    landmark_kernel<<<32, 256, 0, stream>>>(qn, qL, 44, S_TOK);
    landmark_kernel<<<32, 256, 0, stream>>>(kn, kL, 220, TKV);
    attn1_kernel<<<22, 256, 0, stream>>>(qn, kL, a1);
    attn2_kernel<<<128, 256, 0, stream>>>(qL, kn, vc, sc, a2v);
    attn3_kernel<<<704, 512, 0, stream>>>(a1, a2v, vals);

    // out = vals (704,512) x o_w^T + o_b
    gemm_nt_kernel<<<dim3(11, 4), 256, 0, stream>>>(
        vals, o_w, o_b, out, DMODEL, DMODEL);
}
